// SparseNeuralNetwork_19619410608296
// MI455X (gfx1250) — matrix operand order repaired and checked by reading, not compiled, not run on hardware
//
#include <hip/hip_runtime.h>
#include <hip/hip_bf16.h>

// ---------------------------------------------------------------------------
// DenseNet-style skip MLP, fused per-step bf16 WMMA GEMM for gfx1250 (CDNA5).
// bf16 operands / fp32 accumulate (V_WMMA_F32_16X16X32_BF16), with CDNA5
// GLOBAL_LOAD_ASYNC_TO_LDS_B128 (GVS mode) double-buffered tile staging.
// ---------------------------------------------------------------------------

typedef __bf16 bf16_t;
typedef __attribute__((ext_vector_type(4)))  bf16_t v4bf;
typedef __attribute__((ext_vector_type(8)))  bf16_t v8bf;
typedef __attribute__((ext_vector_type(16))) bf16_t v16bf;
typedef __attribute__((ext_vector_type(8)))  float  v8f;

#define BATCH   32768
#define IN_DIM  128
#define WID     256
#define OUT_DIM 64
#define LAYERS  12
#define DEPTH   4

#define BM 128          // workgroup M tile (always)
#define KC 64           // K chunk staged in LDS
#define LDSS 72         // padded LDS row stride (bf16 elems): 144B = 36 banks

struct Branch {
    const bf16_t* A;    // activation  [BATCH x K], bf16, row-major
    const bf16_t* W;    // weight      [N x K],     bf16, row-major (fan_out, fan_in)
    const float*  bias; // [N] fp32 (straight from d_in)
    int K;              // 128 or 256
};

struct StepArgs {
    Branch br[4];
    int    nbr;
    void*  out;         // fp32 (final) or bf16 (hidden) output, row-major
};

// Wave-relative LDS byte offset from a generic pointer to __shared__.
__device__ __forceinline__ uint32_t lds_off(const void* p) {
    return (uint32_t)(uintptr_t)p;
}

// CDNA5 async gather into LDS, GVS mode: per-lane LDS byte offset (VDST),
// per-lane signed 32-bit byte offset (VADDR), uniform 64-bit base (SADDR).
// Tracked with ASYNCcnt.
__device__ __forceinline__ void async_b128(uint32_t lds_byte_off, uint32_t voff_bytes,
                                           const bf16_t* sbase) {
    asm volatile("global_load_async_to_lds_b128 %0, %1, %2"
                 :: "v"(lds_byte_off), "v"(voff_bytes), "s"(sbase)
                 : "memory");
}

#define WAIT_ASYNC(n) asm volatile("s_wait_asynccnt %0" :: "i"(n) : "memory")

// ---------------------------------------------------------------------------
// fp32 -> bf16 conversion (vectorized x4)
// ---------------------------------------------------------------------------
__global__ __launch_bounds__(256)
void cvt_f32_bf16(const float* __restrict__ src, bf16_t* __restrict__ dst, int n4) {
    int i = blockIdx.x * 256 + threadIdx.x;
    if (i < n4) {
        float4 f = reinterpret_cast<const float4*>(src)[i];
        v4bf o;
        o[0] = (bf16_t)f.x; o[1] = (bf16_t)f.y;
        o[2] = (bf16_t)f.z; o[3] = (bf16_t)f.w;
        reinterpret_cast<v4bf*>(dst)[i] = o;
    }
}

// ---------------------------------------------------------------------------
// Fused step kernel: out = sum_b relu(A_b @ W_b^T + bias_b)
// 256 threads = 8 waves laid out 4(M) x 2(N).
//   NTILES = 16-wide WMMA tiles per wave in N; BN = NTILES*32.
//   hidden: NTILES=4 (BN=128, wave tile 32x64), final: NTILES=2 (BN=64).
// Double-buffered LDS tiles filled by global_load_async_to_lds_b128.
// ---------------------------------------------------------------------------
template <int NTILES, bool F32OUT>
__global__ __launch_bounds__(256)
void step_kernel(StepArgs args) {
    constexpr int BN    = NTILES * 32;     // workgroup N tile
    constexpr int WSPAN = NTILES * 16;     // per-wave N span
    constexpr int BITER = BN / 32;         // B-tile async iters per thread
    constexpr int APC   = 4 + BITER;       // async instructions per wave per chunk
    constexpr int NOUT  = F32OUT ? OUT_DIM : WID;

    __shared__ bf16_t As[2 * BM * LDSS];
    __shared__ bf16_t Bs[2 * BN * LDSS];

    const int t    = threadIdx.x;
    const int lane = t & 31;
    const int wave = t >> 5;       // 0..7
    const int wm   = wave & 3;     // M quadrant (32 rows each)
    const int wn   = wave >> 2;    // N half
    const int half = lane >> 4;    // half-wave select (ISA fragment layout)
    const int l15  = lane & 15;

    const int m0 = blockIdx.x * BM;
    const int n0 = blockIdx.y * BN;

    // cooperative-copy indices: 16B per slot, 8 slots per 64-wide row
    const int lr = t >> 3;         // 0..31
    const int lc = (t & 7) * 8;    // element col 0,8,...,56

    const uint32_t asBase = lds_off(As);
    const uint32_t bsBase = lds_off(Bs);

    v8f tot[2][NTILES] = {};       // running branch sum (fp32)

    for (int b = 0; b < args.nbr; ++b) {
        const bf16_t* A    = args.br[b].A;
        const bf16_t* W    = args.br[b].W;
        const float*  bias = args.br[b].bias;
        const int     K    = args.br[b].K;
        const int     nch  = K / KC;      // 2 or 4 (always even)

        // 32-bit per-lane base byte offsets (GVS)
        const uint32_t voffA0 = (uint32_t)(((m0 + lr) * K + lc) * 2);
        const uint32_t voffB0 = (uint32_t)(((n0 + lr) * K + lc) * 2);
        const uint32_t rowStp = (uint32_t)(32 * K * 2);   // 32 rows per iter

        auto issue_chunk = [&](int kc, int buf) {
            const uint32_t kadd = (uint32_t)(kc * 2);
            #pragma unroll
            for (int it = 0; it < 4; ++it)
                async_b128(asBase + (uint32_t)((buf * (BM * LDSS) + (lr + it * 32) * LDSS + lc) * 2),
                           voffA0 + it * rowStp + kadd, A);
            #pragma unroll
            for (int it = 0; it < BITER; ++it)
                async_b128(bsBase + (uint32_t)((buf * (BN * LDSS) + (lr + it * 32) * LDSS + lc) * 2),
                           voffB0 + it * rowStp + kadd, W);
        };

        v8f acc[2][NTILES] = {};   // this branch's GEMM accumulator

        auto compute = [&](int buf) {
            const bf16_t* Ab = &As[buf * (BM * LDSS)];
            const bf16_t* Bb = &Bs[buf * (BN * LDSS)];
            #pragma unroll
            for (int kk = 0; kk < KC; kk += 32) {
                v16bf afrag[2], bfrag[NTILES];
                // A frag: lane holds row M=l15; lo = K[kb..kb+7], hi = K[kb+16..kb+23]
                #pragma unroll
                for (int mt = 0; mt < 2; ++mt) {
                    int r  = wm * 32 + mt * 16 + l15;
                    int kb = kk + half * 8;
                    v8bf lo = *reinterpret_cast<const v8bf*>(&Ab[r * LDSS + kb]);
                    v8bf hi = *reinterpret_cast<const v8bf*>(&Ab[r * LDSS + kb + 16]);
                    afrag[mt] = __builtin_shufflevector(lo, hi,
                        0,1,2,3,4,5,6,7,8,9,10,11,12,13,14,15);
                }
                // B frag: lane holds col N=l15; lo = K[kb..kb+7], hi = K[kb+16..kb+23]
                #pragma unroll
                for (int nt = 0; nt < NTILES; ++nt) {
                    int n  = wn * WSPAN + nt * 16 + l15;
                    int kb = kk + half * 8;
                    v8bf lo = *reinterpret_cast<const v8bf*>(&Bb[n * LDSS + kb]);
                    v8bf hi = *reinterpret_cast<const v8bf*>(&Bb[n * LDSS + kb + 16]);
                    bfrag[nt] = __builtin_shufflevector(lo, hi,
                        0,1,2,3,4,5,6,7,8,9,10,11,12,13,14,15);
                }
                #pragma unroll
                for (int mt = 0; mt < 2; ++mt)
                    #pragma unroll
                    for (int nt = 0; nt < NTILES; ++nt)
                        acc[mt][nt] = __builtin_amdgcn_wmma_f32_16x16x32_bf16(
                            false, afrag[mt], false, bfrag[nt],
                            (short)0, acc[mt][nt], false, false);
            }
        };

        issue_chunk(0, 0);

        // hot loop: unconditional issue-ahead + wait + compute
        int c = 0;
        for (; c < nch - 1; ++c) {
            issue_chunk((c + 1) * KC, (c + 1) & 1);
            WAIT_ASYNC(APC);               // chunk c retired (in-order)
            __syncthreads();               // chunk c visible to all waves
            compute(c & 1);
            __syncthreads();               // all waves done reading buf
        }
        // peeled last chunk
        WAIT_ASYNC(0);
        __syncthreads();
        compute(c & 1);
        __syncthreads();                   // protect buf before next branch issues

        // bias + relu, accumulate into branch sum
        #pragma unroll
        for (int nt = 0; nt < NTILES; ++nt) {
            float bv = bias[n0 + wn * WSPAN + nt * 16 + l15];
            #pragma unroll
            for (int mt = 0; mt < 2; ++mt)
                #pragma unroll
                for (int e = 0; e < 8; ++e) {
                    float v = acc[mt][nt][e] + bv;
                    tot[mt][nt][e] += (v > 0.f) ? v : 0.f;
                }
        }
    }

    // ---- epilogue: compile-time output type & stride ----
    // C/D layout: lane l15 = col, accumulator elem e = row (e + half*8)
    const size_t rbase = (size_t)(m0 + wm * 32 + half * 8) * NOUT
                       + (n0 + wn * WSPAN + l15);
    if (F32OUT) {
        float* base = (float*)args.out + rbase;
        #pragma unroll
        for (int mt = 0; mt < 2; ++mt)
            #pragma unroll
            for (int nt = 0; nt < NTILES; ++nt)
                #pragma unroll
                for (int e = 0; e < 8; ++e)
                    base[(mt * 16 + e) * NOUT + nt * 16] = tot[mt][nt][e];
    } else {
        bf16_t* base = (bf16_t*)args.out + rbase;
        #pragma unroll
        for (int mt = 0; mt < 2; ++mt)
            #pragma unroll
            for (int nt = 0; nt < NTILES; ++nt)
                #pragma unroll
                for (int e = 0; e < 8; ++e)
                    base[(mt * 16 + e) * NOUT + nt * 16] = (bf16_t)tot[mt][nt][e];
    }
}

// ---------------------------------------------------------------------------
// Host side
// ---------------------------------------------------------------------------
extern "C" void kernel_launch(void* const* d_in, const int* in_sizes, int n_in,
                              void* d_out, int out_size, void* d_ws, size_t ws_size,
                              hipStream_t stream) {
    (void)in_sizes; (void)n_in; (void)out_size; (void)ws_size;

    const float* x = (const float*)d_in[0];

    // Flattened param order: for d=1..4 (b=0..3): 13-b layers of (W, bias)
    const float* Wf[4][13];
    const float* Bf[4][13];
    int Kdim[4][13], Ndim[4][13], nlay[4];
    int idx = 1;
    for (int b = 0; b < 4; ++b) {
        int nl = 13 - b;                 // L - (b+1) + 2
        nlay[b] = nl;
        for (int j = 0; j < nl; ++j) {
            Wf[b][j] = (const float*)d_in[idx++];
            Bf[b][j] = (const float*)d_in[idx++];
            Kdim[b][j] = (j == 0) ? IN_DIM : WID;
            Ndim[b][j] = (j == nl - 1) ? OUT_DIM : WID;
        }
    }

    // Workspace carve-up (256B aligned)
    char* ws = (char*)d_ws;
    size_t off = 0;
    auto carve = [&](size_t bytes) -> char* {
        char* p = ws + off;
        off = (off + bytes + 255) & ~(size_t)255;
        return p;
    };

    bf16_t* x_bf = (bf16_t*)carve((size_t)BATCH * IN_DIM * sizeof(bf16_t));
    bf16_t* Wbf[4][13];
    for (int b = 0; b < 4; ++b)
        for (int j = 0; j < nlay[b]; ++j)
            Wbf[b][j] = (bf16_t*)carve((size_t)Ndim[b][j] * Kdim[b][j] * sizeof(bf16_t));
    bf16_t* ring[5];
    for (int s = 0; s < 5; ++s)
        ring[s] = (bf16_t*)carve((size_t)BATCH * WID * sizeof(bf16_t));

    // Convert x and all weights to bf16 (deterministic, every call)
    {
        int n4 = (BATCH * IN_DIM) / 4;
        cvt_f32_bf16<<<(n4 + 255) / 256, 256, 0, stream>>>(x, x_bf, n4);
    }
    for (int b = 0; b < 4; ++b)
        for (int j = 0; j < nlay[b]; ++j) {
            int n4 = (Ndim[b][j] * Kdim[b][j]) / 4;
            cvt_f32_bf16<<<(n4 + 255) / 256, 256, 0, stream>>>(Wf[b][j], Wbf[b][j], n4);
        }

    // 13 fused steps
    for (int i = 0; i <= LAYERS; ++i) {
        StepArgs a = {};
        a.nbr = 0;
        int jlo = (i - (DEPTH - 1) > 0) ? i - (DEPTH - 1) : 0;
        for (int j = jlo; j <= i; ++j) {
            int d = i + 1 - j;           // connection depth, 1..4
            int b = d - 1;
            Branch& br = a.br[a.nbr++];
            br.A    = (j == 0) ? x_bf : ring[(j - 1) % 5];
            br.W    = Wbf[b][j];
            br.bias = Bf[b][j];
            br.K    = Kdim[b][j];
        }
        if (i == LAYERS) {
            a.out = d_out;               // fp32, N = 64, BN = 64
            dim3 grid(BATCH / BM, 1);
            step_kernel<2, true><<<grid, 256, 0, stream>>>(a);
        } else {
            a.out = ring[i % 5];         // bf16, N = 256, BN = 128
            dim3 grid(BATCH / BM, WID / 128);
            step_kernel<4, false><<<grid, 256, 0, stream>>>(a);
        }
    }
}
